// GatedEdgePReLUGNN_33449205301454
// MI455X (gfx1250) — matrix operand order repaired and checked by reading, not compiled, not run on hardware
//
#include <hip/hip_runtime.h>
#include <hip/hip_bf16.h>

typedef __attribute__((ext_vector_type(16))) _Float16 v16h;
typedef __attribute__((ext_vector_type(8)))  float    v8f;

#define EPS_F 1e-5f
#define WMMA_F16(a, b, c) \
  __builtin_amdgcn_wmma_f32_16x16x32_f16(false, (a), false, (b), (short)0, (c), false, false)

// ---------------- fragment helpers ---------------------------------------------
__device__ __forceinline__ v16h ldB(const _Float16* __restrict__ frags, int fragIdx, int lane) {
  return *(const v16h*)(frags + (size_t)fragIdx * 512 + lane * 16);
}

// A fragment from a row-major f32 row: chunks at kOff+8*half and kOff+16+8*half.
__device__ __forceinline__ v16h ldA_row(const float* __restrict__ p, int kOff, int half) {
  const float* p0 = p + kOff + 8 * half;
  float4 u0 = *(const float4*)(p0);
  float4 u1 = *(const float4*)(p0 + 4);
  float4 w0 = *(const float4*)(p0 + 16);
  float4 w1 = *(const float4*)(p0 + 20);
  v16h a;
  a[0] = (_Float16)u0.x; a[1] = (_Float16)u0.y; a[2] = (_Float16)u0.z; a[3] = (_Float16)u0.w;
  a[4] = (_Float16)u1.x; a[5] = (_Float16)u1.y; a[6] = (_Float16)u1.z; a[7] = (_Float16)u1.w;
  a[8]  = (_Float16)w0.x; a[9]  = (_Float16)w0.y; a[10] = (_Float16)w0.z; a[11] = (_Float16)w0.w;
  a[12] = (_Float16)w1.x; a[13] = (_Float16)w1.y; a[14] = (_Float16)w1.z; a[15] = (_Float16)w1.w;
  return a;
}

// ---------------- Utility kernels -----------------------------------------------

__global__ void cvt_frag_kernel(const float* __restrict__ src, int srcRows, int cols,
                                _Float16* __restrict__ dst, int kTiles, int nTiles) {
  int t = blockIdx.x * 256 + threadIdx.x;
  int totalE = kTiles * nTiles * 512;
  if (t >= totalE) return;
  int frag = t >> 9;
  int lane = (t >> 4) & 31;
  int i = t & 15;
  int kt = frag / nTiles, nt = frag % nTiles;
  int k = kt * 32 + 8 * (lane >> 4) + (i & 7) + 16 * (i >> 3);
  int n = nt * 16 + (lane & 15);
  dst[t] = (k < srcRows) ? (_Float16)src[k * cols + n] : (_Float16)0.f;
}

__global__ void init_consts_kernel(float* __restrict__ zeros64, float* __restrict__ ones16) {
  int i = threadIdx.x;
  if (i < 64) zeros64[i] = 0.f;
  if (i < 16) ones16[i] = 1.f;
}

__global__ void deg_kernel(const int* __restrict__ colIdx, float* __restrict__ deg, int E, int N) {
  int e = blockIdx.x * 256 + threadIdx.x;
  if (e >= E + N) return;
  int c = (e < E) ? colIdx[e] : (e - E);
  atomicAdd(deg + c, 1.0f);
}

__global__ void agg_kernel(const int* __restrict__ rowIdx, const int* __restrict__ colIdx,
                           const float* __restrict__ xin, int Cin,
                           float* __restrict__ agg, int E, int N) {
  int e = blockIdx.x * 256 + threadIdx.x;
  if (e >= E + N) return;
  int r = (e < E) ? rowIdx[e] : (e - E);
  int c = (e < E) ? colIdx[e] : (e - E);
  const float4* xr = (const float4*)(xin + (size_t)r * Cin);
  __builtin_prefetch(xr, 0, 0);
  float* ap = agg + (size_t)c * Cin;
#pragma unroll 4
  for (int i = 0; i < Cin / 4; ++i) {
    float4 v = xr[i];
    atomicAdd(ap + 4 * i + 0, v.x);
    atomicAdd(ap + 4 * i + 1, v.y);
    atomicAdd(ap + 4 * i + 2, v.z);
    atomicAdd(ap + 4 * i + 3, v.w);
  }
}

__global__ void scale_by_deg_kernel(float* __restrict__ agg, const float* __restrict__ deg,
                                    int n, int Cin) {
  long long i = (long long)blockIdx.x * 256 + threadIdx.x;
  if (i >= (long long)n * Cin) return;
  agg[i] /= fmaxf(deg[i / Cin], 1.0f);
}

__global__ void reduce_sums_kernel(const float* __restrict__ p1, const float* __restrict__ p2,
                                   int n, float* __restrict__ sums) {
  __shared__ float sS[256], sQ[256];
  int tid = threadIdx.x;
  int c = tid & 63, rgrp = tid >> 6;
  float s = 0.f, q = 0.f;
  for (int i = blockIdx.x * 4 + rgrp; i < n; i += gridDim.x * 4) {
    float v = p1[(size_t)i * 64 + c];
    if (p2) v += p2[(size_t)i * 64 + c];
    s += v; q += v * v;
  }
  sS[tid] = s; sQ[tid] = q;
  __syncthreads();
  if (tid < 64) {
    s = sS[tid] + sS[tid + 64] + sS[tid + 128] + sS[tid + 192];
    q = sQ[tid] + sQ[tid + 64] + sQ[tid + 128] + sQ[tid + 192];
    atomicAdd(&sums[c], s);
    atomicAdd(&sums[64 + c], q);
  }
}

__global__ void finalize_stats_kernel(const float* __restrict__ sums, const float* __restrict__ a,
                                      int n, float* __restrict__ stats) {
  int c = threadIdx.x;
  if (c >= 64) return;
  float inv_n = 1.0f / (float)n;
  float mu = sums[c] * inv_n;
  float m2 = sums[64 + c] * inv_n;
  float av = a ? a[c] : 1.0f;
  float var = m2 - (2.0f * av - av * av) * mu * mu;
  stats[c] = mu;
  stats[64 + c] = rsqrtf(var + EPS_F);
}

__global__ void bn_apply_kernel(const float* __restrict__ out, const float* __restrict__ out2,
                                const float* __restrict__ stats, const float* __restrict__ g,
                                const float* __restrict__ b, float* __restrict__ h, int n) {
  long long i = (long long)blockIdx.x * 256 + threadIdx.x;
  if (i >= (long long)n * 64) return;
  int c = (int)(i & 63);
  float v = out[i] + out2[i];
  v = g[c] * (v - stats[c]) * stats[64 + c] + b[c];
  v *= 2.0f;
  h[i] = v > 0.f ? v : 0.f;
}

__global__ void gn_apply_kernel(float* __restrict__ h, const float* __restrict__ stats,
                                const float* __restrict__ g, const float* __restrict__ b,
                                const float* __restrict__ a, int n) {
  long long i = (long long)blockIdx.x * 256 + threadIdx.x;
  if (i >= (long long)n * 64) return;
  int c = (int)(i & 63);
  float v = h[i];
  v = g[c] * (v - a[c] * stats[c]) * stats[64 + c] + b[c];
  h[i] = v > 0.f ? v : 0.f;
}

// ---------------- WMMA node GEMM (compile-time specialized) ---------------------
// C(n x HOUT) = A(n x K) @ W + bias (+C).  128 threads = 4 waves x 16 rows.
template <int KITERS, int NTILES, bool ACC>
__global__ void gemm_node_kernel(const float* __restrict__ A, int nrows,
                                 const _Float16* __restrict__ Wfrag,
                                 const float* __restrict__ bias,
                                 const float* __restrict__ zeros64,
                                 float* __restrict__ C) {
  constexpr int K = KITERS * 32;
  constexpr int HOUT = NTILES * 16;
  const int lane = threadIdx.x & 31, w = threadIdx.x >> 5;
  const int half = lane >> 4, mloc = lane & 15;
  const int r0 = blockIdx.x * 64 + w * 16;
  const int row = r0 + mloc;
  const float* ap = (row < nrows) ? (A + (size_t)row * K) : zeros64;  // branchless pad

  v16h aF0 = ldA_row(ap, 0, half);
  v16h aF1 = {};
  if (KITERS == 2) aF1 = ldA_row(ap, 32, half);

#pragma unroll
  for (int nt = 0; nt < NTILES; ++nt) {
    v8f acc = {};
    acc = WMMA_F16(aF0, ldB(Wfrag, 0 * NTILES + nt, lane), acc);
    if (KITERS == 2) acc = WMMA_F16(aF1, ldB(Wfrag, 1 * NTILES + nt, lane), acc);
    int n = nt * 16 + mloc;
    float bn = bias[n];
#pragma unroll
    for (int r = 0; r < 8; ++r) {
      int rr = r0 + 8 * half + r;
      if (rr < nrows) {
        size_t idx = (size_t)rr * HOUT + n;
        if (ACC) C[idx] += acc[r] + bn;
        else     C[idx]  = acc[r] + bn;
      }
    }
  }
}

// ---------------- Edge kernel building blocks (static NT) ------------------------

template <int NT>
__device__ __forceinline__ v8f eat_compute(v16h aE, const _Float16* __restrict__ WeF,
                                           const float* __restrict__ be,
                                           int lane, int mloc, int half,
                                           _Float16* __restrict__ sW) {
  v8f c = {};
  c = WMMA_F16(aE, ldB(WeF, NT, lane), c);
  const int n0 = NT * 16 + mloc;
  float ben = be[n0];
#pragma unroll
  for (int r = 0; r < 8; ++r) c[r] += ben;
  // scatter f16 copy into fragment-ordered LDS (gate-A columns 64..127)
  const int kk = 64 + n0;
  const int kt = (kk >> 5) - 2;           // 0 or 1
  const int k_loc = kk & 31;
  const int ha = (k_loc >> 3) & 1;
  const int i = (k_loc & 7) + 8 * (k_loc >> 4);
#pragma unroll
  for (int r = 0; r < 8; ++r) {
    int m = 8 * half + r;
    sW[kt * 512 + (m + 16 * ha) * 16 + i] = (_Float16)c[r];
  }
  return c;
}

template <int NT>
__device__ __forceinline__ void gate_compute(v16h a0, v16h a1, v16h a2, v16h a3,
                                             const _Float16* __restrict__ WgF,
                                             const float* __restrict__ bg, v8f eat,
                                             const int* colR,
                                             float* __restrict__ out2,
                                             float* __restrict__ trash,
                                             int lane, int mloc) {
  v8f acc = {};
  acc = WMMA_F16(a0, ldB(WgF, 0 * 4 + NT, lane), acc);
  acc = WMMA_F16(a1, ldB(WgF, 1 * 4 + NT, lane), acc);
  acc = WMMA_F16(a2, ldB(WgF, 2 * 4 + NT, lane), acc);
  acc = WMMA_F16(a3, ldB(WgF, 3 * 4 + NT, lane), acc);
  const int n = NT * 16 + mloc;
  float bgn = bg[n];
#pragma unroll
  for (int r = 0; r < 8; ++r) {
    float g = 1.0f / (1.0f + __expf(-(acc[r] + bgn)));
    float v = g * eat[r];
    float* dst = (colR[r] >= 0) ? (out2 + (size_t)colR[r] * 64) : trash;  // branchless
    atomicAdd(dst + n, v);
  }
}

// ---------------- Fused per-edge kernel (16 edges per wave) ----------------------
__global__ void edge_kernel(const int* __restrict__ colIdx, const float* __restrict__ ea,
                            const float* __restrict__ out,
                            const _Float16* __restrict__ WeF, const float* __restrict__ be,
                            const _Float16* __restrict__ WgF, const float* __restrict__ bg,
                            const float* __restrict__ zeros64, const float* __restrict__ ones16,
                            float* __restrict__ trash, float* __restrict__ out2,
                            int E, int N) {
  __shared__ _Float16 sEAh[4][1024];

  const int lane = threadIdx.x & 31, w = threadIdx.x >> 5;
  const int half = lane >> 4, mloc = lane & 15;
  const long total = (long)E + N;
  const long e = (long)blockIdx.x * 64 + w * 16 + mloc;

  int myCol = -1;
  if (e < total) myCol = (e < (long)E) ? colIdx[e] : (int)(e - E);

  // ea A fragment (k 0..15 valid, 16..31 zero); self loops -> ones row
  const float* eaRow = (e < (long)E) ? (ea + e * 16) : ((e < total) ? ones16 : zeros64);
  v16h aE = {};
  {
    float4 u0 = *(const float4*)(eaRow + 8 * half);
    float4 u1 = *(const float4*)(eaRow + 8 * half + 4);
    aE[0] = (_Float16)u0.x; aE[1] = (_Float16)u0.y; aE[2] = (_Float16)u0.z; aE[3] = (_Float16)u0.w;
    aE[4] = (_Float16)u1.x; aE[5] = (_Float16)u1.y; aE[6] = (_Float16)u1.z; aE[7] = (_Float16)u1.w;
  }

  _Float16* sW = &sEAh[w][0];
  v8f eat0 = eat_compute<0>(aE, WeF, be, lane, mloc, half, sW);
  v8f eat1 = eat_compute<1>(aE, WeF, be, lane, mloc, half, sW);
  v8f eat2 = eat_compute<2>(aE, WeF, be, lane, mloc, half, sW);
  v8f eat3 = eat_compute<3>(aE, WeF, be, lane, mloc, half, sW);
  __syncthreads();

  const float* outRow = (myCol >= 0) ? (out + (size_t)myCol * 64) : zeros64;
  v16h aG0 = ldA_row(outRow, 0, half);
  v16h aG1 = ldA_row(outRow, 32, half);
  v16h aG2 = *(const v16h*)&sW[lane * 16];
  v16h aG3 = *(const v16h*)&sW[512 + lane * 16];

  int colR[8];
#pragma unroll
  for (int r = 0; r < 8; ++r) colR[r] = __shfl(myCol, 8 * half + r, 32);

  gate_compute<0>(aG0, aG1, aG2, aG3, WgF, bg, eat0, colR, out2, trash, lane, mloc);
  gate_compute<1>(aG0, aG1, aG2, aG3, WgF, bg, eat1, colR, out2, trash, lane, mloc);
  gate_compute<2>(aG0, aG1, aG2, aG3, WgF, bg, eat2, colR, out2, trash, lane, mloc);
  gate_compute<3>(aG0, aG1, aG2, aG3, WgF, bg, eat3, colR, out2, trash, lane, mloc);
}

// ---------------- Host orchestration -------------------------------------------

extern "C" void kernel_launch(void* const* d_in, const int* in_sizes, int n_in,
                              void* d_out, int out_size, void* d_ws, size_t ws_size,
                              hipStream_t stream) {
  const float* x         = (const float*)d_in[0];
  const int*   edge_idx  = (const int*)d_in[1];
  const float* edge_attr = (const float*)d_in[2];
  const float* l1_Wl = (const float*)d_in[3];  const float* l1_bl = (const float*)d_in[4];
  const float* l1_Wr = (const float*)d_in[5];  const float* l1_We = (const float*)d_in[6];
  const float* l1_be = (const float*)d_in[7];  const float* l1_Wg = (const float*)d_in[8];
  const float* l1_bg = (const float*)d_in[9];  const float* l1_bng = (const float*)d_in[10];
  const float* l1_bnb = (const float*)d_in[11];
  const float* l2_Wl = (const float*)d_in[12]; const float* l2_bl = (const float*)d_in[13];
  const float* l2_Wr = (const float*)d_in[14]; const float* l2_We = (const float*)d_in[15];
  const float* l2_be = (const float*)d_in[16]; const float* l2_Wg = (const float*)d_in[17];
  const float* l2_bg = (const float*)d_in[18]; const float* l2_bng = (const float*)d_in[19];
  const float* l2_bnb = (const float*)d_in[20];
  const float* gn1_g = (const float*)d_in[21]; const float* gn1_b = (const float*)d_in[22];
  const float* gn1_a = (const float*)d_in[23];
  const float* gn2_g = (const float*)d_in[24]; const float* gn2_b = (const float*)d_in[25];
  const float* gn2_a = (const float*)d_in[26];
  const float* lin_W = (const float*)d_in[27]; const float* lin_b = (const float*)d_in[28];

  const int N = in_sizes[0] / 32;
  const int E = in_sizes[1] / 2;
  const int total = E + N;
  const int* rowI = edge_idx;
  const int* colI = edge_idx + E;

  char* wp = (char*)d_ws;
  auto carve = [&](size_t bytes) -> char* {
    char* p = wp;
    wp += (bytes + 255) & ~(size_t)255;
    return p;
  };
  float* deg    = (float*)carve((size_t)N * 4);
  float* agg    = (float*)carve((size_t)N * 64 * 4);
  float* outb   = (float*)carve((size_t)N * 64 * 4);
  float* out2   = (float*)carve((size_t)N * 64 * 4);
  float* hbuf   = (float*)carve((size_t)N * 64 * 4);
  float* sums   = (float*)carve(128 * 4);
  float* stats  = (float*)carve(128 * 4);
  float* zeros64 = (float*)carve(64 * 4);
  float* ones16  = (float*)carve(16 * 4);
  float* trash   = (float*)carve(64 * 4);
  _Float16* wWe1 = (_Float16*)carve(1 * 4 * 512 * 2);
  _Float16* wWg1 = (_Float16*)carve(4 * 4 * 512 * 2);
  _Float16* wWl1 = (_Float16*)carve(1 * 4 * 512 * 2);
  _Float16* wWr1 = (_Float16*)carve(1 * 4 * 512 * 2);
  _Float16* wWe2 = (_Float16*)carve(1 * 4 * 512 * 2);
  _Float16* wWg2 = (_Float16*)carve(4 * 4 * 512 * 2);
  _Float16* wWl2 = (_Float16*)carve(2 * 4 * 512 * 2);
  _Float16* wWr2 = (_Float16*)carve(2 * 4 * 512 * 2);
  _Float16* wLin = (_Float16*)carve(2 * 1 * 512 * 2);
  (void)ws_size; (void)n_in; (void)out_size;

  const int eltB = 256;
  auto egrid = [&](long n) { return (int)((n + eltB - 1) / eltB); };

  init_consts_kernel<<<1, 64, 0, stream>>>(zeros64, ones16);

  cvt_frag_kernel<<<egrid(1 * 4 * 512), eltB, 0, stream>>>(l1_We, 16, 64, wWe1, 1, 4);
  cvt_frag_kernel<<<egrid(4 * 4 * 512), eltB, 0, stream>>>(l1_Wg, 128, 64, wWg1, 4, 4);
  cvt_frag_kernel<<<egrid(1 * 4 * 512), eltB, 0, stream>>>(l1_Wl, 32, 64, wWl1, 1, 4);
  cvt_frag_kernel<<<egrid(1 * 4 * 512), eltB, 0, stream>>>(l1_Wr, 32, 64, wWr1, 1, 4);
  cvt_frag_kernel<<<egrid(1 * 4 * 512), eltB, 0, stream>>>(l2_We, 16, 64, wWe2, 1, 4);
  cvt_frag_kernel<<<egrid(4 * 4 * 512), eltB, 0, stream>>>(l2_Wg, 128, 64, wWg2, 4, 4);
  cvt_frag_kernel<<<egrid(2 * 4 * 512), eltB, 0, stream>>>(l2_Wl, 64, 64, wWl2, 2, 4);
  cvt_frag_kernel<<<egrid(2 * 4 * 512), eltB, 0, stream>>>(l2_Wr, 64, 64, wWr2, 2, 4);
  cvt_frag_kernel<<<egrid(2 * 1 * 512), eltB, 0, stream>>>(lin_W, 64, 16, wLin, 2, 1);

  hipMemsetAsync(deg, 0, (size_t)N * 4, stream);
  deg_kernel<<<egrid(total), eltB, 0, stream>>>(colI, deg, E, N);

  const int gemmGrid = (N + 63) / 64;
  const int edgeGrid = (total + 63) / 64;
  const int redGrid = 512;

  // ================= Layer 1 (Cin = 32) =================
  hipMemsetAsync(agg, 0, (size_t)N * 32 * 4, stream);
  hipMemsetAsync(out2, 0, (size_t)N * 64 * 4, stream);
  agg_kernel<<<egrid(total), eltB, 0, stream>>>(rowI, colI, x, 32, agg, E, N);
  scale_by_deg_kernel<<<egrid((long)N * 32), eltB, 0, stream>>>(agg, deg, N, 32);
  gemm_node_kernel<1, 4, false><<<gemmGrid, 128, 0, stream>>>(x, N, wWr1, zeros64, zeros64, outb);
  gemm_node_kernel<1, 4, true><<<gemmGrid, 128, 0, stream>>>(agg, N, wWl1, l1_bl, zeros64, outb);
  edge_kernel<<<edgeGrid, 128, 0, stream>>>(colI, edge_attr, outb, wWe1, l1_be, wWg1, l1_bg,
                                            zeros64, ones16, trash, out2, E, N);
  hipMemsetAsync(sums, 0, 128 * 4, stream);
  reduce_sums_kernel<<<redGrid, 256, 0, stream>>>(outb, out2, N, sums);
  finalize_stats_kernel<<<1, 64, 0, stream>>>(sums, nullptr, N, stats);
  bn_apply_kernel<<<egrid((long)N * 64), eltB, 0, stream>>>(outb, out2, stats, l1_bng, l1_bnb,
                                                            hbuf, N);
  hipMemsetAsync(sums, 0, 128 * 4, stream);
  reduce_sums_kernel<<<redGrid, 256, 0, stream>>>(hbuf, nullptr, N, sums);
  finalize_stats_kernel<<<1, 64, 0, stream>>>(sums, gn1_a, N, stats);
  gn_apply_kernel<<<egrid((long)N * 64), eltB, 0, stream>>>(hbuf, stats, gn1_g, gn1_b, gn1_a, N);

  // ================= Layer 2 (Cin = 64) =================
  hipMemsetAsync(agg, 0, (size_t)N * 64 * 4, stream);
  hipMemsetAsync(out2, 0, (size_t)N * 64 * 4, stream);
  agg_kernel<<<egrid(total), eltB, 0, stream>>>(rowI, colI, hbuf, 64, agg, E, N);
  scale_by_deg_kernel<<<egrid((long)N * 64), eltB, 0, stream>>>(agg, deg, N, 64);
  gemm_node_kernel<2, 4, false><<<gemmGrid, 128, 0, stream>>>(hbuf, N, wWr2, zeros64, zeros64, outb);
  gemm_node_kernel<2, 4, true><<<gemmGrid, 128, 0, stream>>>(agg, N, wWl2, l2_bl, zeros64, outb);
  edge_kernel<<<edgeGrid, 128, 0, stream>>>(colI, edge_attr, outb, wWe2, l2_be, wWg2, l2_bg,
                                            zeros64, ones16, trash, out2, E, N);
  hipMemsetAsync(sums, 0, 128 * 4, stream);
  reduce_sums_kernel<<<redGrid, 256, 0, stream>>>(outb, out2, N, sums);
  finalize_stats_kernel<<<1, 64, 0, stream>>>(sums, nullptr, N, stats);
  bn_apply_kernel<<<egrid((long)N * 64), eltB, 0, stream>>>(outb, out2, stats, l2_bng, l2_bnb,
                                                            hbuf, N);
  hipMemsetAsync(sums, 0, 128 * 4, stream);
  reduce_sums_kernel<<<redGrid, 256, 0, stream>>>(hbuf, nullptr, N, sums);
  finalize_stats_kernel<<<1, 64, 0, stream>>>(sums, gn2_a, N, stats);
  gn_apply_kernel<<<egrid((long)N * 64), eltB, 0, stream>>>(hbuf, stats, gn2_g, gn2_b, gn2_a, N);

  // ================= Final linear =================
  gemm_node_kernel<2, 1, false><<<gemmGrid, 128, 0, stream>>>(hbuf, N, wLin, lin_b, zeros64,
                                                              (float*)d_out);
}
